// LlamaAttention_79517024518576
// MI455X (gfx1250) — compile-verified, matrix operand order checked by reading, not run on hardware
//
#include <hip/hip_runtime.h>
#include <hip/hip_bf16.h>

typedef __attribute__((ext_vector_type(16))) __bf16 v16bf;
typedef __attribute__((ext_vector_type(2)))  __bf16 bf2;
typedef __attribute__((ext_vector_type(8)))  float  v8f;
typedef __attribute__((ext_vector_type(4)))  unsigned int v4u;
typedef __attribute__((ext_vector_type(8)))  int    v8i;
typedef __attribute__((ext_vector_type(4)))  int    v4i;

// Problem sizes (fixed by the reference)
constexpr int Bc  = 2;
constexpr int Tc  = 2048;
constexpr int Hc  = 2048;
constexpr int NHc = 16;
constexpr int HDc = 128;
constexpr int Mc  = Bc * Tc;   // 4096
constexpr int Kc  = Hc;        // 2048
constexpr int Nc  = Hc;        // 2048
constexpr float ATTN_SCALE = 0.08838834764831845f; // 1/sqrt(128)

// A-operand (16x32 bf16) per-lane K-base for vgpr-pair j (ISA 7.12.2 table)
__device__ __forceinline__ int kbase16(int j, int half) {
    return ((j & 3) << 1) + ((j >> 2) << 4) + (half << 3);
}

// ---------------------------------------------------------------------------
// Tensor Data Mover: 2D tile (tile_w x tile_h elements of 2 bytes) from global
// row-major (row stride in elements) into LDS at byte offset lds_off.
// D# packing per CDNA5 ISA 8.3/8.4: count=1, type=2, data_size=1 (2B).
// ---------------------------------------------------------------------------
__device__ __forceinline__ void tdm_load_2d(unsigned lds_off, const void* gptr,
                                            unsigned tile_w, unsigned tile_h,
                                            unsigned long long row_stride) {
    unsigned long long ga = (unsigned long long)(uintptr_t)gptr;
    v4u g0 = { 1u,                                   // count=1, user mode
               lds_off,                              // lds_addr
               (unsigned)ga,                         // global_addr[31:0]
               (unsigned)((ga >> 32) & 0x1FFFFFFu) | (2u << 30) }; // [56:32] | type=2
    v8i g1 = { (int)0x10000,                         // data_size=1 (2 bytes)
               (int)(tile_w << 16),                  // tensor_dim0[15:0]
               (int)((tile_h << 16) | (tile_w >> 16)),   // tdim1 lo | tdim0 hi
               (int)((tile_w << 16) | (tile_h >> 16)),   // tile_dim0 | tdim1 hi
               (int)(tile_h & 0xFFFF),               // tile_dim1 (tile_dim2=0)
               (int)(unsigned)(row_stride & 0xFFFFFFFFull), // dim0_stride lo32
               (int)(unsigned)((row_stride >> 32) & 0xFFFFull), // dim0_stride hi16
               0 };
    v4i gz = { 0, 0, 0, 0 };
#if __clang_major__ >= 23
    v8i gz8 = { 0, 0, 0, 0, 0, 0, 0, 0 };
    __builtin_amdgcn_tensor_load_to_lds(g0, g1, gz, gz, gz8, 0);
#else
    __builtin_amdgcn_tensor_load_to_lds(g0, g1, gz, gz, 0);
#endif
}

// ---------------------------------------------------------------------------
// f32 -> bf16 conversion
// ---------------------------------------------------------------------------
__global__ void cvt_f32_to_bf16(const float* __restrict__ src,
                                __bf16* __restrict__ dst, int n) {
    int i = (blockIdx.x * blockDim.x + threadIdx.x) * 4;
    if (i + 3 < n) {
        float4 f = *reinterpret_cast<const float4*>(src + i);
        dst[i + 0] = (__bf16)f.x;
        dst[i + 1] = (__bf16)f.y;
        dst[i + 2] = (__bf16)f.z;
        dst[i + 3] = (__bf16)f.w;
    } else {
        for (int t = i; t < n; ++t) dst[t] = (__bf16)src[t];
    }
}

// ---------------------------------------------------------------------------
// GEMM: C[M,N] = A[M,K](bf16,row-major) * W[N,K](bf16,row-major)^T
// mode 0: f32 [M,N]            (final output)
// mode 1: bf16 (B,NH,T,HD)     (Q/K fused transpose)
// mode 2: bf16 (B,NH,HD,T)     (V transposed for TDM-friendly attention)
// ---------------------------------------------------------------------------
__global__ __launch_bounds__(256)
void gemm_bf16_wmma(const __bf16* __restrict__ A, const __bf16* __restrict__ W,
                    void* __restrict__ Cout, int mode) {
    __shared__ __align__(16) __bf16 As[128 * 40];
    __shared__ __align__(16) __bf16 Bs[128 * 40];

    const int tid  = threadIdx.x;
    const int lane = tid & 31;
    const int wid  = tid >> 5;
    const int half = lane >> 4;
    const int l16  = lane & 15;
    const int wm   = wid >> 1;
    const int wn   = wid & 1;
    const int nbn  = Nc / 128;
    const int bm   = blockIdx.x / nbn;
    const int bn   = blockIdx.x % nbn;

    const v8f vzero = {0.f, 0.f, 0.f, 0.f, 0.f, 0.f, 0.f, 0.f};
    v8f acc[2][4];
#pragma unroll
    for (int mi = 0; mi < 2; ++mi)
#pragma unroll
        for (int ni = 0; ni < 4; ++ni) acc[mi][ni] = vzero;

    const int ldr = tid >> 2;
    const int ldc = (tid & 3) * 8;

    for (int kc = 0; kc < Kc / 32; ++kc) {
        const int koff = kc * 32;
        if (kc + 1 < Kc / 32) {
            __builtin_prefetch(A + (size_t)(bm * 128 + ldr) * Kc + koff + 32, 0, 0);
            __builtin_prefetch(W + (size_t)(bn * 128 + ldr) * Kc + koff + 32, 0, 0);
        }
#pragma unroll
        for (int p = 0; p < 2; ++p) {
            int r = p * 64 + ldr;
            *reinterpret_cast<uint4*>(&As[r * 40 + ldc]) =
                *reinterpret_cast<const uint4*>(A + (size_t)(bm * 128 + r) * Kc + koff + ldc);
            *reinterpret_cast<uint4*>(&Bs[r * 40 + ldc]) =
                *reinterpret_cast<const uint4*>(W + (size_t)(bn * 128 + r) * Kc + koff + ldc);
        }
        __syncthreads();

        v16bf af[2], bfr[4];
#pragma unroll
        for (int mi = 0; mi < 2; ++mi) {
            int rowl = wm * 32 + mi * 16 + l16;
#pragma unroll
            for (int j = 0; j < 8; ++j) {
                int kb = kbase16(j, half);
                bf2 t2 = *reinterpret_cast<const bf2*>(&As[rowl * 40 + kb]);
                af[mi][2 * j] = t2[0]; af[mi][2 * j + 1] = t2[1];
            }
        }
#pragma unroll
        for (int ni = 0; ni < 4; ++ni) {
            int coll = wn * 64 + ni * 16 + l16;
#pragma unroll
            for (int j = 0; j < 8; ++j) {
                int kb = kbase16(j, half);
                bf2 t2 = *reinterpret_cast<const bf2*>(&Bs[coll * 40 + kb]);
                bfr[ni][2 * j] = t2[0]; bfr[ni][2 * j + 1] = t2[1];
            }
        }
#pragma unroll
        for (int mi = 0; mi < 2; ++mi)
#pragma unroll
            for (int ni = 0; ni < 4; ++ni)
                acc[mi][ni] = __builtin_amdgcn_wmma_f32_16x16x32_bf16(
                    false, af[mi], false, bfr[ni], (short)0, acc[mi][ni], false, false);
        __syncthreads();
    }

#pragma unroll
    for (int mi = 0; mi < 2; ++mi)
#pragma unroll
        for (int ni = 0; ni < 4; ++ni)
#pragma unroll
            for (int r = 0; r < 8; ++r) {
                int gm = bm * 128 + wm * 32 + mi * 16 + r + 8 * half;
                int gn = bn * 128 + wn * 64 + ni * 16 + l16;
                float val = acc[mi][ni][r];
                if (mode == 0) {
                    reinterpret_cast<float*>(Cout)[(size_t)gm * Nc + gn] = val;
                } else {
                    int b = gm >> 11, t = gm & (Tc - 1);
                    int h = gn >> 7,  d = gn & (HDc - 1);
                    size_t idx = (mode == 1)
                        ? (((size_t)(b * NHc + h)) * Tc + t) * HDc + d
                        : (((size_t)(b * NHc + h)) * HDc + d) * Tc + t;
                    reinterpret_cast<__bf16*>(Cout)[idx] = (__bf16)val;
                }
            }
}

// ---------------------------------------------------------------------------
// RoPE on Q and K (bf16, layout B,NH,T,HD), in place
// ---------------------------------------------------------------------------
__global__ void rope_qk(__bf16* __restrict__ q, __bf16* __restrict__ k,
                        const float* __restrict__ cosp, const float* __restrict__ sinp) {
    int idx = blockIdx.x * blockDim.x + threadIdx.x;
    const int total = Bc * NHc * Tc * (HDc / 2);
    if (idx >= total) return;
    int d  = idx & 63;
    int t  = (idx >> 6) & (Tc - 1);
    int bh = idx >> 17;
    size_t base = ((size_t)bh * Tc + t) * HDc;
    float c0 = cosp[t * HDc + d],      s0 = sinp[t * HDc + d];
    float c1 = cosp[t * HDc + d + 64], s1 = sinp[t * HDc + d + 64];

    float q0 = (float)q[base + d], q1 = (float)q[base + d + 64];
    q[base + d]      = (__bf16)(q0 * c0 - q1 * s0);
    q[base + d + 64] = (__bf16)(q1 * c1 + q0 * s1);

    float k0 = (float)k[base + d], k1 = (float)k[base + d + 64];
    k[base + d]      = (__bf16)(k0 * c0 - k1 * s0);
    k[base + d + 64] = (__bf16)(k1 * c1 + k0 * s1);
}

// ---------------------------------------------------------------------------
// Flash attention: 1 wave per block, one 16-row Q tile per wave.
// K (32x128) and V^T (128x32) blocks staged into LDS by the Tensor Data Mover,
// double-buffered on TENSORcnt. K, V and P WMMA operands come from LDS; Q from global.
// Output written bf16 in (B,T,H) layout.
// ---------------------------------------------------------------------------
__global__ __launch_bounds__(32)
void flash_attn_wmma(const __bf16* __restrict__ Q, const __bf16* __restrict__ K,
                     const __bf16* __restrict__ Vt, __bf16* __restrict__ O) {
    __shared__ __align__(16) __bf16 Klds[2][32 * 128];
    __shared__ __align__(16) __bf16 Vlds[2][128 * 32];
    __shared__ __align__(4)  __bf16 Plds[16 * 40];

    const int lane = threadIdx.x & 31;
    const int half = lane >> 4;
    const int l16  = lane & 15;
    const int bh   = blockIdx.x >> 7;          // T/16 = 128 q-tiles per (b,h)
    const int qt   = blockIdx.x & 127;
    const int b    = bh >> 4;
    const int h    = bh & (NHc - 1);
    const int qbase = qt * 16;

    const __bf16* qp  = Q  + (size_t)bh * Tc * HDc;
    const __bf16* kp  = K  + (size_t)bh * Tc * HDc;
    const __bf16* vtp = Vt + (size_t)bh * HDc * Tc;

    // Q fragments: 16 rows x 128 d, split into 4 chunks of K=32
    v16bf qf[4];
#pragma unroll
    for (int dc = 0; dc < 4; ++dc)
#pragma unroll
        for (int j = 0; j < 8; ++j) {
            int kk = kbase16(j, half);
            bf2 t2 = *reinterpret_cast<const bf2*>(
                qp + (size_t)(qbase + l16) * HDc + dc * 32 + kk);
            qf[dc][2 * j] = t2[0]; qf[dc][2 * j + 1] = t2[1];
        }

    const v8f vzero = {0.f, 0.f, 0.f, 0.f, 0.f, 0.f, 0.f, 0.f};
    v8f acc[8];
#pragma unroll
    for (int dt = 0; dt < 8; ++dt) acc[dt] = vzero;
    float mrow[8], lrow[8];
    const float NEG_INF = -__builtin_inff();
#pragma unroll
    for (int r = 0; r < 8; ++r) { mrow[r] = NEG_INF; lrow[r] = 0.f; }

    const int nkb = ((qbase + 15) >> 5) + 1;

    // Prologue: TDM-stage key block 0 into buffer 0
    tdm_load_2d((unsigned)(uintptr_t)(void*)&Klds[0][0], kp, 128, 32, HDc);
    tdm_load_2d((unsigned)(uintptr_t)(void*)&Vlds[0][0], vtp, 32, 128, Tc);

    for (int kb = 0; kb < nkb; ++kb) {
        const int kbb = kb * 32;
        const int buf = kb & 1;
        if (kb + 1 < nkb) {   // stage next block into the other buffer
            const int nb = (kb + 1) & 1;
            tdm_load_2d((unsigned)(uintptr_t)(void*)&Klds[nb][0],
                        kp + (size_t)(kbb + 32) * HDc, 128, 32, HDc);
            tdm_load_2d((unsigned)(uintptr_t)(void*)&Vlds[nb][0],
                        vtp + (size_t)(kbb + 32), 32, 128, Tc);
            __builtin_amdgcn_s_wait_tensorcnt(2);
        } else {
            __builtin_amdgcn_s_wait_tensorcnt(0);
        }
        asm volatile("" ::: "memory");

        // S = Q * K^T (two 16-key subtiles), operands from LDS
        v8f s0 = vzero, s1 = vzero;
#pragma unroll
        for (int dc = 0; dc < 4; ++dc) {
            v16bf bk0, bk1;
#pragma unroll
            for (int j = 0; j < 8; ++j) {
                int kk = kbase16(j, half);
                bf2 t0 = *reinterpret_cast<const bf2*>(&Klds[buf][l16 * 128 + dc * 32 + kk]);
                bf2 t1 = *reinterpret_cast<const bf2*>(&Klds[buf][(16 + l16) * 128 + dc * 32 + kk]);
                bk0[2 * j] = t0[0]; bk0[2 * j + 1] = t0[1];
                bk1[2 * j] = t1[0]; bk1[2 * j + 1] = t1[1];
            }
            s0 = __builtin_amdgcn_wmma_f32_16x16x32_bf16(false, qf[dc], false, bk0,
                                                         (short)0, s0, false, false);
            s1 = __builtin_amdgcn_wmma_f32_16x16x32_bf16(false, qf[dc], false, bk1,
                                                         (short)0, s1, false, false);
        }

        // scale, causal mask, online softmax (rows live in lane-halves)
        float p0[8], p1[8], alpha[8];
#pragma unroll
        for (int r = 0; r < 8; ++r) {
            int row = qbase + r + 8 * half;
            float a0 = (kbb + l16      <= row) ? s0[r] * ATTN_SCALE : NEG_INF;
            float a1 = (kbb + 16 + l16 <= row) ? s1[r] * ATTN_SCALE : NEG_INF;
            float bmx = fmaxf(a0, a1);
            bmx = fmaxf(bmx, __shfl_xor(bmx, 1, 32));
            bmx = fmaxf(bmx, __shfl_xor(bmx, 2, 32));
            bmx = fmaxf(bmx, __shfl_xor(bmx, 4, 32));
            bmx = fmaxf(bmx, __shfl_xor(bmx, 8, 32));
            float mn = fmaxf(mrow[r], bmx);
            alpha[r] = __expf(mrow[r] - mn);
            mrow[r]  = mn;
            p0[r] = __expf(a0 - mn);
            p1[r] = __expf(a1 - mn);
            float rs = p0[r] + p1[r];
            rs += __shfl_xor(rs, 1, 32);
            rs += __shfl_xor(rs, 2, 32);
            rs += __shfl_xor(rs, 4, 32);
            rs += __shfl_xor(rs, 8, 32);
            lrow[r] = lrow[r] * alpha[r] + rs;
        }
#pragma unroll
        for (int dt = 0; dt < 8; ++dt)
#pragma unroll
            for (int r = 0; r < 8; ++r) acc[dt][r] *= alpha[r];

        // Re-layout P (D-layout regs) -> A-fragment via LDS
#pragma unroll
        for (int r = 0; r < 8; ++r) {
            int row = r + 8 * half;
            Plds[row * 40 + l16]      = (__bf16)p0[r];
            Plds[row * 40 + 16 + l16] = (__bf16)p1[r];
        }
        asm volatile("s_wait_dscnt 0" ::: "memory");
        v16bf pa;
#pragma unroll
        for (int j = 0; j < 8; ++j) {
            int kk = kbase16(j, half);
            bf2 t2 = *reinterpret_cast<const bf2*>(&Plds[l16 * 40 + kk]);
            pa[2 * j] = t2[0]; pa[2 * j + 1] = t2[1];
        }

        // O += P * V : V^T tile in LDS gives contiguous key-pairs per lane
#pragma unroll
        for (int dt = 0; dt < 8; ++dt) {
            v16bf bv;
#pragma unroll
            for (int j = 0; j < 8; ++j) {
                int kk = kbase16(j, half);
                bf2 tv = *reinterpret_cast<const bf2*>(&Vlds[buf][(dt * 16 + l16) * 32 + kk]);
                bv[2 * j] = tv[0]; bv[2 * j + 1] = tv[1];
            }
            acc[dt] = __builtin_amdgcn_wmma_f32_16x16x32_bf16(false, pa, false, bv,
                                                              (short)0, acc[dt], false, false);
        }
    }

    float invl[8];
#pragma unroll
    for (int r = 0; r < 8; ++r) invl[r] = 1.0f / lrow[r];
#pragma unroll
    for (int dt = 0; dt < 8; ++dt)
#pragma unroll
        for (int r = 0; r < 8; ++r) {
            int trow = qbase + r + 8 * half;
            O[((size_t)b * Tc + trow) * Hc + h * HDc + dt * 16 + l16] =
                (__bf16)(acc[dt][r] * invl[r]);
        }
}

// ---------------------------------------------------------------------------
extern "C" void kernel_launch(void* const* d_in, const int* in_sizes, int n_in,
                              void* d_out, int out_size, void* d_ws, size_t ws_size,
                              hipStream_t stream) {
    const float* x    = (const float*)d_in[0];
    const float* cosp = (const float*)d_in[1];
    const float* sinp = (const float*)d_in[2];
    const float* Wq   = (const float*)d_in[3];
    const float* Wk   = (const float*)d_in[4];
    const float* Wv   = (const float*)d_in[5];
    const float* Wo   = (const float*)d_in[6];

    const size_t nX = (size_t)Mc * Kc;
    const size_t nW = (size_t)Hc * Hc;
    const size_t nQ = (size_t)Bc * NHc * Tc * HDc;

    char* w = (char*)d_ws;
    __bf16* xb  = (__bf16*)w;                w += nX * 2;
    __bf16* wqb = (__bf16*)w;                w += nW * 2;
    __bf16* wkb = (__bf16*)w;                w += nW * 2;
    __bf16* wvb = (__bf16*)w;                w += nW * 2;
    __bf16* wob = (__bf16*)w;                w += nW * 2;
    __bf16* qb  = (__bf16*)w;                w += nQ * 2;
    __bf16* kb  = (__bf16*)w;                w += nQ * 2;
    __bf16* vb  = (__bf16*)w;                w += nQ * 2;   // stored transposed (B,NH,HD,T)
    __bf16* ob  = (__bf16*)w;                w += nX * 2;

    cvt_f32_to_bf16<<<(int)(nX / 1024), 256, 0, stream>>>(x,  xb,  (int)nX);
    cvt_f32_to_bf16<<<(int)(nW / 1024), 256, 0, stream>>>(Wq, wqb, (int)nW);
    cvt_f32_to_bf16<<<(int)(nW / 1024), 256, 0, stream>>>(Wk, wkb, (int)nW);
    cvt_f32_to_bf16<<<(int)(nW / 1024), 256, 0, stream>>>(Wv, wvb, (int)nW);
    cvt_f32_to_bf16<<<(int)(nW / 1024), 256, 0, stream>>>(Wo, wob, (int)nW);

    const int gemm_grid = (Mc / 128) * (Nc / 128);   // 512
    gemm_bf16_wmma<<<gemm_grid, 256, 0, stream>>>(xb, wqb, qb, 1);
    gemm_bf16_wmma<<<gemm_grid, 256, 0, stream>>>(xb, wkb, kb, 1);
    gemm_bf16_wmma<<<gemm_grid, 256, 0, stream>>>(xb, wvb, vb, 2);

    const int rope_total = Bc * NHc * Tc * (HDc / 2);
    rope_qk<<<rope_total / 256, 256, 0, stream>>>(qb, kb, cosp, sinp);

    flash_attn_wmma<<<Bc * NHc * (Tc / 16), 32, 0, stream>>>(qb, kb, vb, ob);

    gemm_bf16_wmma<<<gemm_grid, 256, 0, stream>>>(ob, wob, d_out, 0);
}
